// CustomVitModel_22239340658968
// MI455X (gfx1250) — compile-verified, matrix operand order checked by reading, not run on hardware
//
#include <hip/hip_runtime.h>

// ---------------- types for WMMA ----------------
typedef __bf16 v16bf __attribute__((ext_vector_type(16)));
typedef float  v8f   __attribute__((ext_vector_type(8)));

union FragAB { v16bf v; uint4 q[2]; };
union FragC  { v8f v; float f[8]; };

// ---------------- model dims ----------------
#define LNUM 12
#define HNUM 12
#define DMOD 768
#define DH   64
#define DFF  3072
#define SEQ  197
#define SPAD 224          // SEQ padded to multiple of 32
#define BATCH 64
#define MTOK (BATCH*SEQ)  // 12608
#define MPAT (BATCH*196)  // 12544
#define NCLS 1000

static __device__ __forceinline__ unsigned short f2bf(float f) {
    unsigned int u = __float_as_uint(f);
    unsigned int r = ((u >> 16) & 1u) + 0x7FFFu;
    return (unsigned short)((u + r) >> 16);
}

// CDNA5 async global->LDS copy, 16B per lane (tracked by ASYNCcnt)
static __device__ __forceinline__ void async_b128(unsigned lds_off, const void* gaddr) {
    asm volatile("global_load_async_to_lds_b128 %0, %1, off"
                 :: "v"(lds_off), "v"(gaddr) : "memory");
}
static __device__ __forceinline__ void wait_async0() {
    asm volatile("s_wait_asynccnt 0" ::: "memory");
}

// =====================================================================
// LDS-staged WMMA GEMM (non-batched): C = alpha*A@Bt^T (+bias)(gelu)(+res)
// Requires M%64==0, N%128==0, K%32==0, lda/ldbt%8==0.
// Block = 8 waves: 4 M-waves x 2 N-waves, block tile 64x128.
// Per K-step the block stages A(64x32) and B(128x32) bf16 tiles into LDS
// via global_load_async_to_lds_b128 (double-buffered), then each wave
// clauses 10 ds_load_b128 fragment reads (80B row stride: 16B aligned,
// conflict-free) and issues 4 back-to-back v_wmma_f32_16x16x32_bf16.
// =====================================================================
__global__ __launch_bounds__(256) void wmma_gemm_lds(
    const unsigned short* __restrict__ A,
    const unsigned short* __restrict__ Bt,
    float* __restrict__ Cf,
    unsigned short* __restrict__ Cb,
    const float* __restrict__ bias,
    const float* __restrict__ res,
    int M, int N, int K,
    int lda, int ldbt, int ldc, int ldcb,
    float alpha, int gelu)
{
    __shared__ unsigned char smem[2 * 64 * 80 + 2 * 128 * 80];   // 30720 B

    const int tid  = threadIdx.x;
    const int lane = tid & 31;
    const int wid  = tid >> 5;
    const int half = lane >> 4;
    const int l16  = lane & 15;

    const int m0 = blockIdx.x * 64 + (wid & 3) * 16;
    const int n0 = blockIdx.y * 128 + (wid >> 2) * 64;

    unsigned char* aT = smem;                    // [2][64][80]
    unsigned char* bT = smem + 2 * 64 * 80;      // [2][128][80]

    // cooperative copy mapping: 16B per thread per issue
    const int irow = tid >> 2;                   // 0..63
    const int icol = (tid & 3) * 16;             // byte offset in 64B row

    const char* gA  = (const char*)A  + ((long)(blockIdx.x * 64 + irow) * lda) * 2 + icol;
    const char* gB0 = (const char*)Bt + ((long)(blockIdx.y * 128 + irow) * ldbt) * 2 + icol;
    const char* gB1 = (const char*)Bt + ((long)(blockIdx.y * 128 + 64 + irow) * ldbt) * 2 + icol;

    auto issue = [&](int buf, int k0) {
        const long kb = (long)k0 * 2;
        async_b128((unsigned)(size_t)(aT + buf * 64 * 80 + irow * 80 + icol), gA + kb);
        async_b128((unsigned)(size_t)(bT + buf * 128 * 80 + irow * 80 + icol), gB0 + kb);
        async_b128((unsigned)(size_t)(bT + buf * 128 * 80 + (64 + irow) * 80 + icol), gB1 + kb);
    };

    FragC acc[4];
#pragma unroll
    for (int t = 0; t < 4; ++t)
#pragma unroll
        for (int r = 0; r < 8; ++r) acc[t].f[r] = 0.0f;

    issue(0, 0);
    wait_async0();
    __syncthreads();

    for (int k0 = 0; k0 < K; k0 += 32) {
        const int buf = (k0 >> 5) & 1;
        if (k0 + 32 < K) issue(buf ^ 1, k0 + 32);   // prefetch next K-step

        // clause all fragment loads, then back-to-back WMMAs
        FragAB a, b[4];
        const unsigned char* ap = aT + buf * 64 * 80 + ((wid & 3) * 16 + l16) * 80 + half * 16;
        a.q[0] = *(const uint4*)(ap);
        a.q[1] = *(const uint4*)(ap + 32);
#pragma unroll
        for (int t = 0; t < 4; ++t) {
            const unsigned char* bp = bT + buf * 128 * 80 +
                                      ((wid >> 2) * 64 + t * 16 + l16) * 80 + half * 16;
            b[t].q[0] = *(const uint4*)(bp);
            b[t].q[1] = *(const uint4*)(bp + 32);
        }
#pragma unroll
        for (int t = 0; t < 4; ++t)
            acc[t].v = __builtin_amdgcn_wmma_f32_16x16x32_bf16(
                false, a.v, false, b[t].v, (short)0, acc[t].v, false, false);

        wait_async0();        // next buffer landed
        __syncthreads();      // everyone done reading current buffer
    }

    // ---------------- epilogue (full tiles, no guards) ----------------
#pragma unroll
    for (int t = 0; t < 4; ++t) {
        const int nc = n0 + t * 16 + l16;
        const float bv = bias ? bias[nc] : 0.0f;
#pragma unroll
        for (int r = 0; r < 8; ++r) {
            const int m = m0 + r + half * 8;
            float v = acc[t].f[r] * alpha + bv;
            if (gelu) v = 0.5f * v * (1.0f + erff(v * 0.70710678118f));
            if (res)  v += res[(long)m * ldc + nc];
            if (Cf)   Cf[(long)m * ldc  + nc] = v;
            if (Cb)   Cb[(long)m * ldcb + nc] = f2bf(v);
        }
    }
}

// =====================================================================
// Direct-load WMMA GEMM (batched): GUARDED=false needs M%64==0 && N%64==0.
// =====================================================================
template<bool GUARDED>
__global__ __launch_bounds__(256) void wmma_gemm_t(
    const unsigned short* __restrict__ A,
    const unsigned short* __restrict__ Bt,
    float* __restrict__ Cf,
    unsigned short* __restrict__ Cb,
    const float* __restrict__ bias,
    const float* __restrict__ res,
    int M, int N, int K,
    int lda, int ldbt, int ldc, int ldcb,
    long sA0, long sA1, long sB0, long sB1, long sC0, long sC1, long sBias1,
    int bdim1, float alpha, int gelu)
{
    const int lane = threadIdx.x & 31;
    const int wid  = threadIdx.x >> 5;
    const int half = lane >> 4;
    const int l16  = lane & 15;

    const int b0 = blockIdx.z / bdim1;
    const int b1 = blockIdx.z % bdim1;
    A  += b0 * sA0 + b1 * sA1;
    Bt += b0 * sB0 + b1 * sB1;
    const long coff = b0 * sC0 + b1 * sC1;
    if (bias) bias += b1 * sBias1;

    const int m0 = blockIdx.x * 64 + (wid & 3) * 16;
    const int n0 = blockIdx.y * 128 + (wid >> 2) * 64;
    if (m0 >= M || n0 >= N) return;   // wave-uniform exit

    const int arow = m0 + l16;

    FragC acc[4];
#pragma unroll
    for (int t = 0; t < 4; ++t)
#pragma unroll
        for (int r = 0; r < 8; ++r) acc[t].f[r] = 0.0f;

    if (!GUARDED) {
        const unsigned short* ap  = A + (long)arow * lda + half * 8;
        const unsigned short* bp0 = Bt + (long)(n0 + l16) * ldbt + half * 8;
#pragma unroll 2
        for (int k0 = 0; k0 < K; k0 += 32) {
            FragAB a, b[4];
            a.q[0] = *(const uint4*)(ap);
            a.q[1] = *(const uint4*)(ap + 16);
            __builtin_prefetch(ap + 32, 0, 1);
#pragma unroll
            for (int t = 0; t < 4; ++t) {
                const unsigned short* bp = bp0 + (long)t * 16 * ldbt;
                b[t].q[0] = *(const uint4*)(bp);
                b[t].q[1] = *(const uint4*)(bp + 16);
            }
#pragma unroll
            for (int t = 0; t < 4; ++t)
                acc[t].v = __builtin_amdgcn_wmma_f32_16x16x32_bf16(
                    false, a.v, false, b[t].v, (short)0, acc[t].v, false, false);
            ap  += 32;
            bp0 += 32;
        }
    } else {
        const bool aOK = arow < M;
        uint4 z; z.x = z.y = z.z = z.w = 0u;
        for (int k0 = 0; k0 < K; k0 += 32) {
            FragAB a;
            const unsigned short* ap = A + (long)arow * lda + k0 + half * 8;
            if (aOK) { a.q[0] = *(const uint4*)(ap); a.q[1] = *(const uint4*)(ap + 16); }
            else     { a.q[0] = z; a.q[1] = z; }
#pragma unroll
            for (int t = 0; t < 4; ++t) {
                if (n0 + t * 16 >= N) continue;          // wave-uniform
                const int  ncol = n0 + t * 16 + l16;
                const bool bOK  = ncol < N;
                FragAB b;
                const unsigned short* bp = Bt + (long)ncol * ldbt + k0 + half * 8;
                if (bOK) { b.q[0] = *(const uint4*)(bp); b.q[1] = *(const uint4*)(bp + 16); }
                else     { b.q[0] = z; b.q[1] = z; }
                acc[t].v = __builtin_amdgcn_wmma_f32_16x16x32_bf16(
                    false, a.v, false, b.v, (short)0, acc[t].v, false, false);
            }
        }
    }

    // ---------------- epilogue ----------------
#pragma unroll
    for (int t = 0; t < 4; ++t) {
        if (GUARDED && n0 + t * 16 >= N) continue;
        const int nc = n0 + t * 16 + l16;
        if (GUARDED && nc >= N) continue;
        const float bv = bias ? bias[nc] : 0.0f;
#pragma unroll
        for (int r = 0; r < 8; ++r) {
            const int m = m0 + r + half * 8;
            if (GUARDED && m >= M) continue;
            float v = acc[t].f[r] * alpha + bv;
            if (gelu) v = 0.5f * v * (1.0f + erff(v * 0.70710678118f));
            if (res)  v += res[coff + (long)m * ldc + nc];
            if (Cf)   Cf[coff + (long)m * ldc  + nc] = v;
            if (Cb)   Cb[coff + (long)m * ldcb + nc] = f2bf(v);
        }
    }
}

// ------------- batched transpose + fp32->bf16 convert: out[b][c][r] = in[b][r][c]
__global__ void k_transpose_bf16(const float* __restrict__ in, unsigned short* __restrict__ out,
                                 int R, int C, int total)
{
    int idx = blockIdx.x * 256 + threadIdx.x;
    if (idx >= total) return;
    int rc = R * C;
    int b = idx / rc, rm = idx % rc;
    int r = rm / C, c = rm % C;
    out[(long)b * rc + (long)c * R + r] = f2bf(in[idx]);
}

// ------------- patchify: images[B,3,224,224] -> bf16 [B*196, 768]
__global__ void k_patchify(const float* __restrict__ img, unsigned short* __restrict__ out, int total)
{
    int idx = blockIdx.x * 256 + threadIdx.x;
    if (idx >= total) return;
    int col = idx % 768, r = idx / 768;
    int n = r / 196, p = r % 196;
    int c = col >> 8, rem = col & 255;
    int py = rem >> 4, px = rem & 15;
    int gy = (p / 14) * 16 + py, gx = (p % 14) * 16 + px;
    out[idx] = f2bf(img[(((long)n * 3 + c) * 224 + gy) * 224 + gx]);
}

// ------------- assemble tokens + cls + sinusoidal pos-emb -> out[B,197,768] fp32
__global__ void k_assemble(const float* __restrict__ tok, const float* __restrict__ cls,
                           float* __restrict__ out, int total)
{
    int idx = blockIdx.x * 256 + threadIdx.x;
    if (idx >= total) return;
    int d = idx % 768, r = idx / 768;
    int n = r / SEQ, s = r % SEQ;
    float expo = (float)(d & ~1) * (1.0f / 768.0f);
    float ang  = (float)s * __powf(10000.0f, -expo);
    float pe   = (d & 1) ? __cosf(ang) : __sinf(ang);
    float base = (s == 0) ? cls[d] : tok[((long)n * 196 + s - 1) * 768 + d];
    out[idx] = base + pe;
}

// ------------- LayerNorm per row (width=768) -> bf16
__global__ __launch_bounds__(256) void k_layernorm(const float* __restrict__ x,
                                                   const float* __restrict__ g,
                                                   const float* __restrict__ b,
                                                   unsigned short* __restrict__ out, int width)
{
    int row = blockIdx.x;
    const float* xr = x + (long)row * width;
    float s1 = 0.f, s2 = 0.f;
    for (int c = threadIdx.x; c < width; c += 256) { float v = xr[c]; s1 += v; s2 += v * v; }
    __shared__ float r1[256], r2[256];
    r1[threadIdx.x] = s1; r2[threadIdx.x] = s2;
    __syncthreads();
    for (int o = 128; o > 0; o >>= 1) {
        if (threadIdx.x < o) { r1[threadIdx.x] += r1[threadIdx.x + o]; r2[threadIdx.x] += r2[threadIdx.x + o]; }
        __syncthreads();
    }
    float mean = r1[0] / width;
    float var  = r2[0] / width - mean * mean;
    float inv  = rsqrtf(var + 1e-5f);
    for (int c = threadIdx.x; c < width; c += 256)
        out[(long)row * width + c] = f2bf((xr[c] - mean) * inv * g[c] + b[c]);
}

// ------------- row softmax (1 wave / row); optional bf16 out with zeroed pad cols
__global__ __launch_bounds__(256) void k_softmax(const float* __restrict__ in, int ld,
                                                 int Wd, int P,
                                                 unsigned short* __restrict__ outb,
                                                 float* __restrict__ outf, int ldo, int rows)
{
    int row = blockIdx.x * 8 + (threadIdx.x >> 5);
    if (row >= rows) return;
    int lane = threadIdx.x & 31;
    const float* xr = in + (long)row * ld;
    float m = -3.4e38f;
    for (int c = lane; c < Wd; c += 32) m = fmaxf(m, xr[c]);
    for (int o = 16; o; o >>= 1) m = fmaxf(m, __shfl_xor(m, o, 32));
    float s = 0.f;
    for (int c = lane; c < Wd; c += 32) s += __expf(xr[c] - m);
    for (int o = 16; o; o >>= 1) s += __shfl_xor(s, o, 32);
    float inv = 1.0f / s;
    if (outb) {
        unsigned short* orow = outb + (long)row * ldo;
        for (int c = lane; c < Wd; c += 32) orow[c] = f2bf(__expf(xr[c] - m) * inv);
        for (int c = Wd + lane; c < P; c += 32) orow[c] = 0;
    }
    if (outf) {
        float* orow = outf + (long)row * ldo;
        for (int c = lane; c < Wd; c += 32) orow[c] = __expf(xr[c] - m) * inv;
    }
}

// ------------- v[B*S,768] bf16 -> vT[B,H,64,SPAD] bf16 (zero pad s>=SEQ)
__global__ void k_vtrans(const unsigned short* __restrict__ v, unsigned short* __restrict__ vT, int total)
{
    int idx = blockIdx.x * 256 + threadIdx.x;
    if (idx >= total) return;
    int s = idx % SPAD, rem = idx / SPAD;
    int e = rem % DH, rem2 = rem / DH;   // rem2 = n*H + h
    int h = rem2 % HNUM, n = rem2 / HNUM;
    vT[idx] = (s < SEQ) ? v[((long)n * SEQ + s) * DMOD + h * DH + e] : (unsigned short)0;
}

// ------------- extract cls token rows -> bf16 [64,768]
__global__ void k_cls(const float* __restrict__ out, unsigned short* __restrict__ cb, int total)
{
    int idx = blockIdx.x * 256 + threadIdx.x;
    if (idx >= total) return;
    int n = idx / DMOD, d = idx % DMOD;
    cb[idx] = f2bf(out[(long)n * SEQ * DMOD + d]);
}

// =====================================================================
static inline int cdiv(int a, int b) { return (a + b - 1) / b; }

static void gemm(hipStream_t st, const unsigned short* A, const unsigned short* Bt,
                 float* Cf, unsigned short* Cb, const float* bias, const float* res,
                 int M, int N, int K, int lda, int ldbt, int ldc, int ldcb,
                 long sA0, long sA1, long sB0, long sB1, long sC0, long sC1, long sBias1,
                 int bdim1, int nbatch, float alpha, int gelu)
{
    dim3 grid(cdiv(M, 64), cdiv(N, 128), nbatch);
    if (nbatch == 1 && (M % 64 == 0) && (N % 128 == 0))
        wmma_gemm_lds<<<grid, 256, 0, st>>>(A, Bt, Cf, Cb, bias, res, M, N, K,
                                            lda, ldbt, ldc, ldcb, alpha, gelu);
    else if ((M % 64 == 0) && (N % 64 == 0))
        wmma_gemm_t<false><<<grid, 256, 0, st>>>(A, Bt, Cf, Cb, bias, res, M, N, K,
                                                 lda, ldbt, ldc, ldcb,
                                                 sA0, sA1, sB0, sB1, sC0, sC1, sBias1,
                                                 bdim1, alpha, gelu);
    else
        wmma_gemm_t<true><<<grid, 256, 0, st>>>(A, Bt, Cf, Cb, bias, res, M, N, K,
                                                lda, ldbt, ldc, ldcb,
                                                sA0, sA1, sB0, sB1, sC0, sC1, sBias1,
                                                bdim1, alpha, gelu);
}

extern "C" void kernel_launch(void* const* d_in, const int* in_sizes, int n_in,
                              void* d_out, int out_size, void* d_ws, size_t ws_size,
                              hipStream_t stream)
{
    const float* images = (const float*)d_in[0];
    const float* Wp     = (const float*)d_in[1];
    const float* bp     = (const float*)d_in[2];
    const float* cls    = (const float*)d_in[3];
    const float* ln1_g  = (const float*)d_in[4];
    const float* ln1_b  = (const float*)d_in[5];
    const float* Wq     = (const float*)d_in[6];
    const float* bq     = (const float*)d_in[7];
    const float* Wk     = (const float*)d_in[8];
    const float* bk     = (const float*)d_in[9];
    const float* Wv     = (const float*)d_in[10];
    const float* bv     = (const float*)d_in[11];
    const float* ln2_g  = (const float*)d_in[12];
    const float* ln2_b  = (const float*)d_in[13];
    const float* W1     = (const float*)d_in[14];
    const float* b1     = (const float*)d_in[15];
    const float* W2     = (const float*)d_in[16];
    const float* b2     = (const float*)d_in[17];
    const float* Wh     = (const float*)d_in[18];
    const float* bh     = (const float*)d_in[19];

    // ---- workspace carve-up ----
    size_t off = 0;
    auto alloc = [&](size_t elems, size_t elsize) -> void* {
        void* p = (char*)d_ws + off;
        off = (off + elems * elsize + 255) & ~(size_t)255;
        return p;
    };
    unsigned short* wTp = (unsigned short*)alloc((size_t)768 * 768, 2);
    unsigned short* wTq = (unsigned short*)alloc((size_t)LNUM * HNUM * DH * DH, 2);
    unsigned short* wTk = (unsigned short*)alloc((size_t)LNUM * HNUM * DH * DH, 2);
    unsigned short* wTv = (unsigned short*)alloc((size_t)LNUM * HNUM * DH * DH, 2);
    unsigned short* wT1 = (unsigned short*)alloc((size_t)LNUM * DMOD * DFF, 2);
    unsigned short* wT2 = (unsigned short*)alloc((size_t)LNUM * DFF * DMOD, 2);
    unsigned short* wTh = (unsigned short*)alloc((size_t)NCLS * DMOD, 2);
    unsigned short* patches = (unsigned short*)alloc((size_t)MPAT * 768, 2);
    unsigned short* h_bf  = (unsigned short*)alloc((size_t)MTOK * DMOD, 2);
    unsigned short* q_bf  = (unsigned short*)alloc((size_t)MTOK * DMOD, 2);
    unsigned short* k_bf  = (unsigned short*)alloc((size_t)MTOK * DMOD, 2);
    unsigned short* v_bf  = (unsigned short*)alloc((size_t)MTOK * DMOD, 2);
    unsigned short* vT    = (unsigned short*)alloc((size_t)BATCH * HNUM * DH * SPAD, 2);
    unsigned short* attn  = (unsigned short*)alloc((size_t)BATCH * HNUM * SEQ * SPAD, 2);
    unsigned short* mid   = (unsigned short*)alloc((size_t)MTOK * DFF, 2);
    unsigned short* clsb  = (unsigned short*)alloc((size_t)BATCH * DMOD, 2);
    float* tokens = (float*)alloc((size_t)MPAT * 768, 4);
    float* out    = (float*)alloc((size_t)MTOK * DMOD, 4);
    float* scores = (float*)alloc((size_t)BATCH * HNUM * SEQ * SPAD, 4);
    float* logits = (float*)alloc((size_t)BATCH * NCLS, 4);

    // ---- weight transpose+convert (fp32 -> bf16 Bt layouts) ----
    {
        int t;
        t = 768 * 768;
        k_transpose_bf16<<<cdiv(t, 256), 256, 0, stream>>>(Wp, wTp, 768, 768, t);
        t = LNUM * HNUM * DH * DH;
        k_transpose_bf16<<<cdiv(t, 256), 256, 0, stream>>>(Wq, wTq, DH, DH, t);
        k_transpose_bf16<<<cdiv(t, 256), 256, 0, stream>>>(Wk, wTk, DH, DH, t);
        k_transpose_bf16<<<cdiv(t, 256), 256, 0, stream>>>(Wv, wTv, DH, DH, t);
        t = LNUM * DMOD * DFF;
        k_transpose_bf16<<<cdiv(t, 256), 256, 0, stream>>>(W1, wT1, DMOD, DFF, t);
        k_transpose_bf16<<<cdiv(t, 256), 256, 0, stream>>>(W2, wT2, DFF, DMOD, t);
        t = DMOD * NCLS;
        k_transpose_bf16<<<cdiv(t, 256), 256, 0, stream>>>(Wh, wTh, DMOD, NCLS, t);
    }

    // ---- patch embed ----
    {
        int t = MPAT * 768;
        k_patchify<<<cdiv(t, 256), 256, 0, stream>>>(images, patches, t);
        gemm(stream, patches, wTp, tokens, nullptr, bp, nullptr,
             MPAT, 768, 768, 768, 768, 768, 0,
             0, 0, 0, 0, 0, 0, 0, 1, 1, 1.0f, 0);
        t = MTOK * DMOD;
        k_assemble<<<cdiv(t, 256), 256, 0, stream>>>(tokens, cls, out, t);
    }

    const long sQKV = (long)SEQ * DMOD;        // per-image stride in token matrices
    const long sSC  = (long)SEQ * SPAD;        // per-(n,h) stride in scores/attn
    const long sVT  = (long)DH * SPAD;         // per-(n,h) stride in vT

    for (int l = 0; l < LNUM; ++l) {
        // LN1 -> h_bf
        k_layernorm<<<MTOK, 256, 0, stream>>>(out, ln1_g + (size_t)l * DMOD,
                                              ln1_b + (size_t)l * DMOD, h_bf, DMOD);
        // q/k/v projections: batch over heads (A col-offset h*64, Bt per-head 64x64)
        gemm(stream, h_bf, wTq + (size_t)l * HNUM * DH * DH, nullptr, q_bf,
             bq + (size_t)l * HNUM * DH, nullptr,
             MTOK, DH, DH, DMOD, DH, DMOD, DMOD,
             0, DH, 0, (long)DH * DH, 0, DH, DH, HNUM, HNUM, 1.0f, 0);
        gemm(stream, h_bf, wTk + (size_t)l * HNUM * DH * DH, nullptr, k_bf,
             bk + (size_t)l * HNUM * DH, nullptr,
             MTOK, DH, DH, DMOD, DH, DMOD, DMOD,
             0, DH, 0, (long)DH * DH, 0, DH, DH, HNUM, HNUM, 1.0f, 0);
        gemm(stream, h_bf, wTv + (size_t)l * HNUM * DH * DH, nullptr, v_bf,
             bv + (size_t)l * HNUM * DH, nullptr,
             MTOK, DH, DH, DMOD, DH, DMOD, DMOD,
             0, DH, 0, (long)DH * DH, 0, DH, DH, HNUM, HNUM, 1.0f, 0);
        // v -> vT [B,H,64,SPAD] (zero-padded K for attn@V)
        {
            int t = BATCH * HNUM * DH * SPAD;
            k_vtrans<<<cdiv(t, 256), 256, 0, stream>>>(v_bf, vT, t);
        }
        // scores = q @ k^T * scale   (batch B*H; b0=image, b1=head)
        gemm(stream, q_bf, k_bf, scores, nullptr, nullptr, nullptr,
             SEQ, SEQ, DH, DMOD, DMOD, SPAD, 0,
             sQKV, DH, sQKV, DH, (long)HNUM * sSC, sSC, 0,
             HNUM, BATCH * HNUM, 0.125f, 0);
        // softmax -> attn (bf16, pad cols zeroed)
        {
            int rows = BATCH * HNUM * SEQ;
            k_softmax<<<cdiv(rows, 8), 256, 0, stream>>>(scores, SPAD, SEQ, SPAD,
                                                         attn, nullptr, SPAD, rows);
        }
        // out += attn @ v   (residual fused into epilogue)
        gemm(stream, attn, vT, out, nullptr, nullptr, out,
             SEQ, DH, SPAD, SPAD, SPAD, DMOD, 0,
             (long)HNUM * sSC, sSC, (long)HNUM * sVT, sVT, sQKV, DH, 0,
             HNUM, BATCH * HNUM, 1.0f, 0);
        // LN2 -> h_bf
        k_layernorm<<<MTOK, 256, 0, stream>>>(out, ln2_g + (size_t)l * DMOD,
                                              ln2_b + (size_t)l * DMOD, h_bf, DMOD);
        // mid = gelu(h_bf @ W1 + b1)  (bf16 out)
        gemm(stream, h_bf, wT1 + (size_t)l * DMOD * DFF, nullptr, mid,
             b1 + (size_t)l * DFF, nullptr,
             MTOK, DFF, DMOD, DMOD, DMOD, DFF, DFF,
             0, 0, 0, 0, 0, 0, 0, 1, 1, 1.0f, 1);
        // out += mid @ W2 + b2
        gemm(stream, mid, wT2 + (size_t)l * DFF * DMOD, out, nullptr,
             b2 + (size_t)l * DMOD, out,
             MTOK, DMOD, DFF, DFF, DFF, DMOD, 0,
             0, 0, 0, 0, 0, 0, 0, 1, 1, 1.0f, 0);
    }

    // ---- head: logits = out[:,0] @ Wh + bh ; softmax -> d_out ----
    {
        int t = BATCH * DMOD;
        k_cls<<<cdiv(t, 256), 256, 0, stream>>>(out, clsb, t);
        gemm(stream, clsb, wTh, logits, nullptr, bh, nullptr,
             BATCH, NCLS, DMOD, DMOD, DMOD, NCLS, 0,
             0, 0, 0, 0, 0, 0, 0, 1, 1, 1.0f, 0);
        k_softmax<<<cdiv(BATCH, 8), 256, 0, stream>>>(logits, NCLS, NCLS, NCLS,
                                                      nullptr, (float*)d_out, NCLS, BATCH);
    }
    (void)in_sizes; (void)n_in; (void)out_size; (void)ws_size;
}
